// AttentionLayer_57114475102265
// MI455X (gfx1250) — matrix operand order repaired and checked by reading, not compiled, not run on hardware
//
#include <hip/hip_runtime.h>

// ---------------------------------------------------------------------------
// Problem constants (from reference): B=4, L=S=2048, D=1024, H=16, E=64
// ---------------------------------------------------------------------------
#define BB 4
#define LL 2048
#define SS 2048
#define DD 1024
#define HH 16
#define EE 64

typedef __attribute__((ext_vector_type(16))) __bf16 v16bf;
typedef __attribute__((ext_vector_type(8)))  __bf16 v8bf;
typedef __attribute__((ext_vector_type(4)))  __bf16 v4bf;
typedef __attribute__((ext_vector_type(8)))  float  v8f;
typedef __attribute__((ext_vector_type(4)))  unsigned int u32x4;
typedef __attribute__((ext_vector_type(8)))  int  i32x8;
typedef __attribute__((ext_vector_type(4)))  int  i32x4;

union AF { v16bf v; v8bf h2[2]; };

static __device__ __forceinline__ v8f wmma_bf16(const v16bf& a, const v16bf& b, const v8f& c) {
    return __builtin_amdgcn_wmma_f32_16x16x32_bf16(false, a, false, b, (short)0, c, false, false);
}

// ---------------------------------------------------------------------------
// TDM: issue a 2-D tensor DMA of a 128-row x 32-col bf16 tile into LDS.
// D# built per CDNA5 ISA ch.8: group0 = {ctl, lds_addr, gaddr_lo, gaddr_hi|type},
// group1 = {data_size, dims, tile dims, strides}; groups 2/3 unused (2-D).
// ---------------------------------------------------------------------------
static __device__ __forceinline__ void tdm_load_tile_128x32(
    const __bf16* gsrc, unsigned lds_byte_addr, unsigned tensor_rows)
{
    const unsigned long long ga = (unsigned long long)(const void*)gsrc;
    u32x4 g0;
    g0[0] = 1u;                                              // count=1 (valid), user mode
    g0[1] = lds_byte_addr;                                   // LDS dest (bytes)
    g0[2] = (unsigned)ga;                                    // global_addr[31:0]
    g0[3] = (unsigned)((ga >> 32) & 0x01ffffffu)             // global_addr[56:32]
          | 0x80000000u;                                     // type=2 ("image")
    i32x8 g1;
    g1[0] = 0x00010000;                                      // data_size=1 -> 2 bytes
    g1[1] = (int)(1024u << 16);                              // tensor_dim0 = 1024 (lo16)
    g1[2] = (int)(tensor_rows << 16);                        // dim0 hi16=0 | tensor_dim1 lo16
    g1[3] = (int)((tensor_rows >> 16) | (32u << 16));        // tensor_dim1 hi16 | tile_dim0=32
    g1[4] = 128;                                             // tile_dim1=128, tile_dim2=0
    g1[5] = 1024;                                            // tensor_dim0_stride lo32 = 1024
    g1[6] = 0;                                               // stride hi16 | dim1_stride lo16
    g1[7] = 0;                                               // dim1_stride hi32
    i32x4 z4 = {0, 0, 0, 0};
#if __clang_major__ >= 23
    i32x8 z8 = {0, 0, 0, 0, 0, 0, 0, 0};
    __builtin_amdgcn_tensor_load_to_lds(g0, g1, z4, z4, z8, 0);
#else
    __builtin_amdgcn_tensor_load_to_lds(g0, g1, z4, z4, 0);
#endif
}

// ---------------------------------------------------------------------------
// fp32 -> bf16 elementwise conversion (vectorized x4)
// ---------------------------------------------------------------------------
__global__ __launch_bounds__(256)
void f32_to_bf16_kernel(const float* __restrict__ src, __bf16* __restrict__ dst)
{
    const size_t i = ((size_t)blockIdx.x * 256 + threadIdx.x) * 4;
    const float4 f = *(const float4*)(src + i);
    v4bf o;
    o[0] = (__bf16)f.x; o[1] = (__bf16)f.y; o[2] = (__bf16)f.z; o[3] = (__bf16)f.w;
    *(v4bf*)(dst + i) = o;
}

// ---------------------------------------------------------------------------
// GEMM: out[m][n] = sum_k X[m][k] * W[n][k] + bias[n]   (X, W pre-converted bf16)
// M = 8192, N = K = 1024. Block tile 128x128, 8 waves (2x4), wave tile 64x32.
// Tiles staged by the Tensor Data Mover, double-buffered in LDS.
// ---------------------------------------------------------------------------
template <typename TOUT>
__global__ __launch_bounds__(256)
void gemm_bf16(const __bf16* __restrict__ X, const __bf16* __restrict__ Wb,
               const float* __restrict__ bias, TOUT* __restrict__ out)
{
    constexpr int K = DD, N = DD, KSTEPS = K / 32;
    __shared__ __bf16 ldsA[2][128 * 32];
    __shared__ __bf16 ldsB[2][128 * 32];

    const int tid   = threadIdx.x;
    const int lane  = tid & 31;
    const int w     = tid >> 5;
    const int waveM = w >> 2;        // 0..1
    const int waveN = w & 3;         // 0..3
    const int half  = lane >> 4;     // 0/1
    const int l15   = lane & 15;
    const int bm    = blockIdx.y * 128;
    const int bn    = blockIdx.x * 128;

    const __bf16* Abase = X  + (size_t)bm * K;
    const __bf16* Bbase = Wb + (size_t)bn * K;

    // prologue: DMA tile 0 into buffer 0 (one wave drives the TDM)
    if (w == 0) {
        tdm_load_tile_128x32(Abase, (unsigned)(unsigned long long)&ldsA[0][0], 8192u);
        tdm_load_tile_128x32(Bbase, (unsigned)(unsigned long long)&ldsB[0][0], 1024u);
    }

    v8f acc[4][2] = {};

    for (int kt = 0; kt < KSTEPS; ++kt) {
        const int cur = kt & 1;
        const int nxt = cur ^ 1;
        if (w == 0) {
            if (kt + 1 < KSTEPS) {
                // prefetch next tile into the other buffer, then wait for the
                // oldest two transfers (current buffer) to land: TENSORcnt <= 2
                tdm_load_tile_128x32(Abase + (kt + 1) * 32,
                                     (unsigned)(unsigned long long)&ldsA[nxt][0], 8192u);
                tdm_load_tile_128x32(Bbase + (kt + 1) * 32,
                                     (unsigned)(unsigned long long)&ldsB[nxt][0], 1024u);
                __builtin_amdgcn_s_wait_tensorcnt(2);
            } else {
                __builtin_amdgcn_s_wait_tensorcnt(0);
            }
        }
        __syncthreads();

        // A fragments: row m = lane&15, 16 contiguous K at base=16*half
        AF a[4];
        #pragma unroll
        for (int mi = 0; mi < 4; ++mi) {
            const __bf16* p = &ldsA[cur][(waveM * 64 + mi * 16 + l15) * 32 + half * 16];
            a[mi].h2[0] = *(const v8bf*)(p);
            a[mi].h2[1] = *(const v8bf*)(p + 8);
        }
        // B fragments: col n = lane&15, 16 contiguous K at base=16*half
        AF b[2];
        #pragma unroll
        for (int ni = 0; ni < 2; ++ni) {
            const __bf16* p = &ldsB[cur][(waveN * 32 + ni * 16 + l15) * 32 + half * 16];
            b[ni].v = *(const v16bf*)(p);
        }
        #pragma unroll
        for (int mi = 0; mi < 4; ++mi)
            #pragma unroll
            for (int ni = 0; ni < 2; ++ni)
                acc[mi][ni] = wmma_bf16(a[mi].v, b[ni].v, acc[mi][ni]);

        __syncthreads();
    }

    // store: C layout -> (m = i + 8*half, n = lane&15)
    #pragma unroll
    for (int mi = 0; mi < 4; ++mi) {
        #pragma unroll
        for (int ni = 0; ni < 2; ++ni) {
            const int col = bn + waveN * 32 + ni * 16 + l15;
            const float bv = bias[col];
            #pragma unroll
            for (int i = 0; i < 8; ++i) {
                const int row = bm + waveM * 64 + mi * 16 + i + half * 8;
                out[(size_t)row * N + col] = (TOUT)(acc[mi][ni][i] + bv);
            }
        }
    }
}

// ---------------------------------------------------------------------------
// vh [B*S][D] -> vhT [(b*H+h)*E + e][S]   (makes P@V B-fragments contiguous)
// ---------------------------------------------------------------------------
__global__ __launch_bounds__(256)
void transpose_v(const __bf16* __restrict__ vh, __bf16* __restrict__ vhT)
{
    const int s   = blockIdx.x * 256 + threadIdx.x;      // 0..2047
    const int row = blockIdx.y;                           // 0..B*H*E-1
    const int e   = row & (EE - 1);
    const int bh  = row >> 6;
    const int h   = bh & (HH - 1);
    const int b   = bh >> 4;
    vhT[(size_t)row * SS + s] = vh[((size_t)(b * SS + s)) * DD + h * EE + e];
}

// ---------------------------------------------------------------------------
// Flash-style attention: grid (L/128, B*H), 8 waves, wave owns 16 query rows.
// logits = (scale*tau)*QK^T + scale*delta[s]; online softmax; O = P@V.
// K/V fragments streamed straight from global (all activations are L2-resident).
// ---------------------------------------------------------------------------
__global__ __launch_bounds__(256)
void attention_kernel(const __bf16* __restrict__ qh, const __bf16* __restrict__ kh,
                      const __bf16* __restrict__ vhT, const float* __restrict__ tau,
                      const float* __restrict__ delta, __bf16* __restrict__ ctx)
{
    __shared__ __bf16 ldsP[8][16 * 64];

    const int w    = threadIdx.x >> 5;
    const int lane = threadIdx.x & 31;
    const int half = lane >> 4;
    const int l15  = lane & 15;
    const int bh   = blockIdx.y;
    const int h    = bh & (HH - 1);
    const int b    = bh >> 4;
    const int q0   = blockIdx.x * 128 + w * 16;          // first of 16 query rows
    const float scale = 0.125f;                           // 1/sqrt(64)
    const float st = tau[b] * scale;

    // Q fragments (E=64 -> two K=32 chunks), held for the whole S loop
    AF qa[2];
    {
        const __bf16* qrow = qh + ((size_t)(b * LL) + q0 + l15) * DD + h * EE;
        #pragma unroll
        for (int c = 0; c < 2; ++c) {
            const __bf16* p = qrow + c * 32 + half * 16;
            qa[c].h2[0] = *(const v8bf*)(p);
            qa[c].h2[1] = *(const v8bf*)(p + 8);
        }
    }

    v8f o[4] = {};
    float mrow[8], lrow[8];
    #pragma unroll
    for (int i = 0; i < 8; ++i) { mrow[i] = -3.0e38f; lrow[i] = 0.0f; }

    const __bf16* kbase = kh + ((size_t)b * SS) * DD + h * EE;
    const __bf16* vbase = vhT + ((size_t)bh * EE) * SS;
    const float*  dbase = delta + (size_t)b * SS;

    for (int s0 = 0; s0 < SS; s0 += 64) {
        // ---- scores: 16x64 tile = 4 j-tiles, each 2 WMMAs over E ----
        v8f sc[4];
        #pragma unroll
        for (int j = 0; j < 4; ++j) {
            v8f accv = {};
            #pragma unroll
            for (int c = 0; c < 2; ++c) {
                AF kb;  // B frag: col n = s, 16 contiguous e
                kb.v = *(const v16bf*)(kbase + (size_t)(s0 + j * 16 + l15) * DD + c * 32 + half * 16);
                accv = wmma_bf16(qa[c].v, kb.v, accv);
            }
            sc[j] = accv;
        }

        // ---- logits + row max (reduce across the 16-lane half) ----
        float dj[4];
        #pragma unroll
        for (int j = 0; j < 4; ++j) dj[j] = dbase[s0 + j * 16 + l15] * scale;

        float alpha[8];
        #pragma unroll
        for (int i = 0; i < 8; ++i) {
            float m = -3.0e38f;
            #pragma unroll
            for (int j = 0; j < 4; ++j) {
                sc[j][i] = st * sc[j][i] + dj[j];
                m = fmaxf(m, sc[j][i]);
            }
            #pragma unroll
            for (int mk = 1; mk < 16; mk <<= 1) m = fmaxf(m, __shfl_xor(m, mk, 32));
            const float mn = fmaxf(mrow[i], m);
            alpha[i] = __expf(mrow[i] - mn);
            mrow[i]  = mn;
        }

        // ---- p = exp(logit - m), row sums, online l update ----
        #pragma unroll
        for (int i = 0; i < 8; ++i) {
            float rs = 0.0f;
            #pragma unroll
            for (int j = 0; j < 4; ++j) {
                const float p = __expf(sc[j][i] - mrow[i]);
                sc[j][i] = p;
                rs += p;
            }
            #pragma unroll
            for (int mk = 1; mk < 16; mk <<= 1) rs += __shfl_xor(rs, mk, 32);
            lrow[i] = lrow[i] * alpha[i] + rs;
        }

        // ---- rescale O ----
        #pragma unroll
        for (int t = 0; t < 4; ++t)
            #pragma unroll
            for (int i = 0; i < 8; ++i) o[t][i] *= alpha[i];

        // ---- P (C layout) -> LDS -> A-layout fragments (wave-private region) ----
        #pragma unroll
        for (int j = 0; j < 4; ++j)
            #pragma unroll
            for (int i = 0; i < 8; ++i)
                ldsP[w][(i + half * 8) * 64 + j * 16 + l15] = (__bf16)sc[j][i];

        AF pa[2];
        #pragma unroll
        for (int c = 0; c < 2; ++c) {
            const __bf16* p = &ldsP[w][l15 * 64 + c * 32 + half * 16];
            pa[c].h2[0] = *(const v8bf*)(p);
            pa[c].h2[1] = *(const v8bf*)(p + 8);
        }

        // ---- O += P @ V : 4 e-tiles x 2 K-chunks ----
        #pragma unroll
        for (int t = 0; t < 4; ++t) {
            #pragma unroll
            for (int c = 0; c < 2; ++c) {
                AF vb;  // B frag: col n = e, 16 contiguous s in vhT
                vb.v = *(const v16bf*)(vbase + (size_t)(t * 16 + l15) * SS + s0 + c * 32 + half * 16);
                o[t] = wmma_bf16(pa[c].v, vb.v, o[t]);
            }
        }
    }

    // ---- finalize: O /= l, store bf16 context [B*L][D] ----
    #pragma unroll
    for (int t = 0; t < 4; ++t) {
        #pragma unroll
        for (int i = 0; i < 8; ++i) {
            const int row = b * LL + q0 + i + half * 8;
            const float val = o[t][i] / lrow[i];
            ctx[(size_t)row * DD + h * EE + t * 16 + l15] = (__bf16)val;
        }
    }
}

// ---------------------------------------------------------------------------
// Host-side launcher
// ---------------------------------------------------------------------------
extern "C" void kernel_launch(void* const* d_in, const int* in_sizes, int n_in,
                              void* d_out, int out_size, void* d_ws, size_t ws_size,
                              hipStream_t stream)
{
    (void)in_sizes; (void)n_in; (void)out_size; (void)ws_size;

    const float* q     = (const float*)d_in[0];
    const float* k     = (const float*)d_in[1];
    const float* v     = (const float*)d_in[2];
    const float* tau   = (const float*)d_in[3];
    const float* delta = (const float*)d_in[4];
    const float* Wq    = (const float*)d_in[5];
    const float* bq    = (const float*)d_in[6];
    const float* Wk    = (const float*)d_in[7];
    const float* bk    = (const float*)d_in[8];
    const float* Wv    = (const float*)d_in[9];
    const float* bv    = (const float*)d_in[10];
    const float* Wo    = (const float*)d_in[11];
    const float* bo    = (const float*)d_in[12];
    float* out = (float*)d_out;

    const size_t NE = (size_t)BB * LL * DD;   // 8,388,608 activation elements
    const size_t NW = (size_t)DD * DD;        // 1,048,576 weight elements
    __bf16* ws   = (__bf16*)d_ws;
    __bf16* qh   = ws;
    __bf16* kh   = ws + NE;
    __bf16* vh   = ws + 2 * NE;
    __bf16* vhT  = ws + 3 * NE;
    __bf16* ctx  = ws + 4 * NE;
    __bf16* Xq   = ws + 5 * NE;
    __bf16* Xk   = ws + 6 * NE;
    __bf16* Xv   = ws + 7 * NE;
    __bf16* WqB  = ws + 8 * NE;
    __bf16* WkB  = WqB + NW;
    __bf16* WvB  = WqB + 2 * NW;
    __bf16* WoB  = WqB + 3 * NW;

    dim3 blk(256);
    dim3 gGemm(DD / 128, (BB * LL) / 128);    // (8, 64)
    const int gAct = (int)(NE / 1024);        // convert grid: activations
    const int gWgt = (int)(NW / 1024);        // convert grid: weights

    // fp32 -> bf16 staging copies (enables TDM DMA + halves GEMM read traffic)
    f32_to_bf16_kernel<<<gAct, blk, 0, stream>>>(q,  Xq);
    f32_to_bf16_kernel<<<gAct, blk, 0, stream>>>(k,  Xk);
    f32_to_bf16_kernel<<<gAct, blk, 0, stream>>>(v,  Xv);
    f32_to_bf16_kernel<<<gWgt, blk, 0, stream>>>(Wq, WqB);
    f32_to_bf16_kernel<<<gWgt, blk, 0, stream>>>(Wk, WkB);
    f32_to_bf16_kernel<<<gWgt, blk, 0, stream>>>(Wv, WvB);
    f32_to_bf16_kernel<<<gWgt, blk, 0, stream>>>(Wo, WoB);

    gemm_bf16<__bf16><<<gGemm, blk, 0, stream>>>(Xq, WqB, bq, qh);
    gemm_bf16<__bf16><<<gGemm, blk, 0, stream>>>(Xk, WkB, bk, kh);
    gemm_bf16<__bf16><<<gGemm, blk, 0, stream>>>(Xv, WvB, bv, vh);

    transpose_v<<<dim3(SS / 256, BB * HH * EE), blk, 0, stream>>>(vh, vhT);

    attention_kernel<<<dim3(LL / 128, BB * HH), blk, 0, stream>>>(qh, kh, vhT, tau, delta, ctx);

    gemm_bf16<float><<<gGemm, blk, 0, stream>>>(ctx, WoB, bo, out);
}
